// RecipeGenModel_40200893891296
// MI455X (gfx1250) — matrix operand order repaired and checked by reading, not compiled, not run on hardware
//
#include <hip/hip_runtime.h>
#include <math.h>

// ---------------- problem constants ----------------
#define BB 32
#define SS 100
#define TT 100
#define EE 128
#define HH 256
#define GG 768   /* 3*H */
#define VV 32000

typedef __bf16 bf16;
typedef __attribute__((ext_vector_type(16))) __bf16 v16bf;
typedef __attribute__((ext_vector_type(8)))  float  v8f;

// ---------------- WMMA fragment helpers (layouts per CDNA5 ISA 7.12.2) ----
// A (16x32 bf16): lane L<16 -> M=L, K = {k0..k0+7, k0+16..k0+23};
//                 lane L>=16 -> M=L-16, K halves offset by +8.
__device__ __forceinline__ v16bf load_a16(const bf16* Ab, int lda, int m0, int k0, int lane) {
  const int m  = m0 + (lane & 15);
  const int kb = k0 + ((lane & 16) ? 8 : 0);
  const bf16* p = Ab + (size_t)m * lda + kb;
  v16bf f;
#pragma unroll
  for (int i = 0; i < 8; ++i) { f[i] = p[i]; f[8 + i] = p[16 + i]; }
  return f;
}
// B (32x16 bf16) from row-major (N,K) weights: lane -> column n = n0+(lane&15),
// K = {kb..kb+7, kb+16..kb+23}, kb = k0 + (lane<16 ? 0 : 8) (matches A). No transpose needed.
__device__ __forceinline__ v16bf load_b16(const bf16* Wb, int ldw, int n0, int k0, int lane) {
  const int n  = n0 + (lane & 15);
  const int kl = k0 + ((lane & 16) ? 8 : 0);
  const bf16* p = Wb + (size_t)n * ldw + kl;
  v16bf f;
#pragma unroll
  for (int i = 0; i < 8; ++i) { f[i] = p[i]; f[8 + i] = p[16 + i]; }
  return f;
}
__device__ __forceinline__ v8f wmma_bf16(v16bf a, v16bf b, v8f c) {
  return __builtin_amdgcn_wmma_f32_16x16x32_bf16(false, a, false, b, (short)0, c, false, false);
}
// C/D (16x16 f32): lane -> N = lane&15; VGPR r -> M = r + (lane<16 ? 0 : 8).
__device__ __forceinline__ void store_c(float* C, int ldc, int m0, int n0, int lane, v8f acc) {
  const int n  = n0 + (lane & 15);
  const int mb = m0 + ((lane & 16) ? 8 : 0);
#pragma unroll
  for (int r = 0; r < 8; ++r) C[(mb + r) * ldc + n] = acc[r];
}
__device__ __forceinline__ float sigmoidf_(float x) { return 1.0f / (1.0f + expf(-x)); }

// ---------------- small utility kernels ----------------
__global__ void k_cast_f32_bf16(const float* __restrict__ s, bf16* __restrict__ d, int n) {
  int i = blockIdx.x * blockDim.x + threadIdx.x;
  if (i < n) d[i] = (bf16)s[i];
}

__global__ void k_gather_src(const int* __restrict__ ing, const float* __restrict__ emb,
                             bf16* __restrict__ x_enc) {
  int i = blockIdx.x * blockDim.x + threadIdx.x;   // over B*S*E
  if (i >= BB * SS * EE) return;
  int e = i & (EE - 1);
  int bs = i >> 7;                                  // E = 128
  x_enc[i] = (bf16)emb[(size_t)ing[bs] * EE + e];
}

// ---------------- encoder: persistent single-WGP GRU, WMMA gates ----------
// 512 threads = 16 waves (4/SIMD -> 256 VGPRs/wave: no spills).
// LDS (dynamic): gi[32][768] f32, gh[32][768] f32, hb[32][256] bf16
__global__ void __launch_bounds__(512)
k_enc_gru(const bf16* __restrict__ x_enc,   // (B,S,E) bf16
          const bf16* __restrict__ Wih,     // (768,128) bf16
          const bf16* __restrict__ Whh,     // (768,256) bf16
          const float* __restrict__ bih, const float* __restrict__ bhh,
          float* __restrict__ enc_outs,     // (B,S,H) f32
          float* __restrict__ d_h, bf16* __restrict__ d_hb) {
  extern __shared__ char smem[];
  float* gi = (float*)smem;                // 32*768
  float* gh = gi + BB * GG;                // 32*768
  bf16*  hb = (bf16*)(gh + BB * GG);       // 32*256

  const int tid  = threadIdx.x;
  const int lane = tid & 31;
  const int wave = tid >> 5;               // 16 waves

  float hreg[16];                           // h state: thread owns i = tid + r*512
#pragma unroll
  for (int r = 0; r < 16; ++r) hreg[r] = 0.0f;
  for (int i = tid; i < BB * HH; i += 512) hb[i] = (bf16)0.0f;

  for (int s = 0; s < SS; ++s) {
    __syncthreads();                        // hb from previous combine ready; gi/gh consumed
    // gi = x_s @ Wih^T : 96 tiles (2m x 48n), K=128; 6 tiles/wave, rolled (uniform branch)
#pragma unroll 1
    for (int tt = 0; tt < 6; ++tt) {
      const int t  = wave + tt * 16;
      const int mt = t & 1, nt = t >> 1;
      v8f acc = {};
#pragma unroll
      for (int k0 = 0; k0 < EE; k0 += 32) {
        v16bf a = load_a16(x_enc + (size_t)s * EE, SS * EE, mt * 16, k0, lane);
        v16bf b = load_b16(Wih, EE, nt * 16, k0, lane);
        acc = wmma_bf16(a, b, acc);
      }
      store_c(gi, GG, mt * 16, nt * 16, lane, acc);
    }
    // gh = h @ Whh^T : K=256 (A from LDS bf16 h)
#pragma unroll 1
    for (int tt = 0; tt < 6; ++tt) {
      const int t  = wave + tt * 16;
      const int mt = t & 1, nt = t >> 1;
      v8f acc = {};
#pragma unroll
      for (int k0 = 0; k0 < HH; k0 += 32) {
        v16bf a = load_a16(hb, HH, mt * 16, k0, lane);
        v16bf b = load_b16(Whh, HH, nt * 16, k0, lane);
        acc = wmma_bf16(a, b, acc);
      }
      store_c(gh, GG, mt * 16, nt * 16, lane, acc);
    }
    __syncthreads();                        // gates ready
    // combine: GRU nonlinearity + state update
#pragma unroll
    for (int r = 0; r < 16; ++r) {
      int i = tid + r * 512;                // i in [0, 8192)
      int b = i >> 8, j = i & 255;
      float gir = gi[b * GG + j]        + bih[j];
      float giz = gi[b * GG + HH + j]   + bih[HH + j];
      float gin = gi[b * GG + 2*HH + j] + bih[2*HH + j];
      float ghr = gh[b * GG + j]        + bhh[j];
      float ghz = gh[b * GG + HH + j]   + bhh[HH + j];
      float ghn = gh[b * GG + 2*HH + j] + bhh[2*HH + j];
      float rg = sigmoidf_(gir + ghr);
      float zg = sigmoidf_(giz + ghz);
      float ng = tanhf(gin + rg * ghn);
      float hn = (1.0f - zg) * ng + zg * hreg[r];
      hreg[r] = hn;
      hb[i]   = (bf16)hn;
      enc_outs[((size_t)b * SS + s) * HH + j] = hn;
      if (s == SS - 1) { d_h[i] = hn; d_hb[i] = (bf16)hn; }
    }
  }
}

// ---------------- attention (constant across decoder steps!) --------------
// scores = enc_part + (h@w_h) per-row constant => softmax shift-invariant =>
// w = softmax(enc_part) and ctx are step-invariant. Compute once.
__global__ void k_attn(const float* __restrict__ enc_outs, const float* __restrict__ attn_W,
                       const float* __restrict__ attn_b, bf16* __restrict__ ctx_bf) {
  __shared__ float sc[SS + 1];
  const int b = blockIdx.x, tid = threadIdx.x;   // 32 blocks x 128 threads
  if (tid < SS) {
    const float* row = enc_outs + ((size_t)b * SS + tid) * HH;
    float d = 0.0f;
    for (int k = 0; k < HH; ++k) d += row[k] * attn_W[k];
    sc[tid] = d + attn_b[0];
  }
  __syncthreads();
  if (tid == 0) {
    float mx = sc[0];
    for (int s = 1; s < SS; ++s) mx = fmaxf(mx, sc[s]);
    float sum = 0.0f;
    for (int s = 0; s < SS; ++s) { float e = expf(sc[s] - mx); sc[s] = e; sum += e; }
    sc[SS] = 1.0f / sum;
  }
  __syncthreads();
  const float inv = sc[SS];
  for (int h = tid; h < HH; h += blockDim.x) {
    float c = 0.0f;
    for (int s = 0; s < SS; ++s) c += sc[s] * enc_outs[((size_t)b * SS + s) * HH + h];
    ctx_bf[b * HH + h] = (bf16)(c * inv);
  }
}

// ---------------- decoder GRU step (single WGP, WMMA) ----------------------
// Fused head: argmax-combine of previous step's partials -> tok (t==0 -> tok=1).
// 512 threads = 16 waves. LDS: gi, gh (f32), x[32][384] bf16, tok[32] int
__global__ void __launch_bounds__(512)
k_dec_gru(const bf16* __restrict__ emb_bf, const bf16* __restrict__ ctx_bf,
          const float* __restrict__ part_val, const int* __restrict__ part_idx,
          const bf16* __restrict__ Wih,    // (768,384)
          const bf16* __restrict__ Whh,    // (768,256)
          const float* __restrict__ bih, const float* __restrict__ bhh,
          float* __restrict__ d_h, bf16* __restrict__ d_hb, int t) {
  extern __shared__ char smem[];
  float* gi = (float*)smem;
  float* gh = gi + BB * GG;
  bf16*  x  = (bf16*)(gh + BB * GG);               // 32*384
  int* tok_s = (int*)(x + BB * (EE + HH));         // 32

  const int tid = threadIdx.x, lane = tid & 31, wave = tid >> 5;
  const int K1 = EE + HH;                           // 384

  // ---- fused argmax: reduce 125 per-block partials (ascending => first max, jnp.argmax) ----
  if (tid < BB) {
    int bi = 1;                                     // tok0 = ones
    if (t > 0) {
      float best = -INFINITY; bi = 0;
      for (int blk = 0; blk < VV / 256; ++blk) {
        float v = part_val[blk * BB + tid];
        if (v > best) { best = v; bi = part_idx[blk * BB + tid]; }
      }
    }
    tok_s[tid] = bi;
  }
  __syncthreads();

  // ---- stage x = [emb[tok], ctx] ----
  for (int i = tid; i < BB * K1; i += 512) {
    int b = i / K1, j = i - b * K1;
    x[i] = (j < EE) ? emb_bf[(size_t)tok_s[b] * EE + j] : ctx_bf[b * HH + (j - EE)];
  }
  __syncthreads();

#pragma unroll 1
  for (int tt = 0; tt < 6; ++tt) {                  // gi = x @ Wih^T, K=384
    const int t2 = wave + tt * 16;
    const int mt = t2 & 1, nt = t2 >> 1;
    v8f acc = {};
#pragma unroll
    for (int k0 = 0; k0 < 384; k0 += 32) {
      v16bf a = load_a16(x, K1, mt * 16, k0, lane);
      v16bf b = load_b16(Wih, K1, nt * 16, k0, lane);
      acc = wmma_bf16(a, b, acc);
    }
    store_c(gi, GG, mt * 16, nt * 16, lane, acc);
  }
#pragma unroll 1
  for (int tt = 0; tt < 6; ++tt) {                  // gh = h @ Whh^T, K=256 (A from global bf16 h)
    const int t2 = wave + tt * 16;
    const int mt = t2 & 1, nt = t2 >> 1;
    v8f acc = {};
#pragma unroll
    for (int k0 = 0; k0 < HH; k0 += 32) {
      v16bf a = load_a16(d_hb, HH, mt * 16, k0, lane);
      v16bf b = load_b16(Whh, HH, nt * 16, k0, lane);
      acc = wmma_bf16(a, b, acc);
    }
    store_c(gh, GG, mt * 16, nt * 16, lane, acc);
  }
  __syncthreads();
#pragma unroll
  for (int r = 0; r < 16; ++r) {
    int i = tid + r * 512;
    int b = i >> 8, j = i & 255;
    float gir = gi[b * GG + j]        + bih[j];
    float giz = gi[b * GG + HH + j]   + bih[HH + j];
    float gin = gi[b * GG + 2*HH + j] + bih[2*HH + j];
    float ghr = gh[b * GG + j]        + bhh[j];
    float ghz = gh[b * GG + HH + j]   + bhh[HH + j];
    float ghn = gh[b * GG + 2*HH + j] + bhh[2*HH + j];
    float rg = sigmoidf_(gir + ghr);
    float zg = sigmoidf_(giz + ghz);
    float ng = tanhf(gin + rg * ghn);
    float hn = (1.0f - zg) * ng + zg * d_h[i];
    d_h[i]  = hn;
    d_hb[i] = (bf16)hn;
  }
}

// ---------------- output projection: pred = h2 @ out_W^T + out_b ----------
// 125 blocks x 256 threads (8 waves, 2/SIMD -> 512 VGPRs/wave); 256 vocab cols/block.
// B fragment shared across both M tiles (out_W read once); prefetch next n-tile.
// Fused: bias add, store to d_out[b][t][v], per-block partial argmax.
__global__ void __launch_bounds__(256)
k_out_gemm(const bf16* __restrict__ hb, const bf16* __restrict__ outW,
           const float* __restrict__ out_b, float* __restrict__ out,
           int t, float* __restrict__ part_val, int* __restrict__ part_idx) {
  __shared__ float preds[BB * 256];
  const int tid = threadIdx.x, lane = tid & 31, wave = tid >> 5;
  const int nbase = blockIdx.x * 256;

  // prefetch second n-tile's B rows into cache while first tile computes
  {
    const bf16* pn = outW + (size_t)(nbase + (wave + 8) * 16 + (lane & 15)) * HH;
    __builtin_prefetch(pn, 0, 0);
  }
#pragma unroll
  for (int q = 0; q < 2; ++q) {                     // 2 n-tiles per wave (uniform)
    const int nt = wave + q * 8;
    v8f acc0 = {}, acc1 = {};
#pragma unroll
    for (int k0 = 0; k0 < HH; k0 += 32) {
      v16bf bfrag = load_b16(outW, HH, nbase + nt * 16, k0, lane);
      v16bf a0 = load_a16(hb, HH, 0,  k0, lane);
      v16bf a1 = load_a16(hb, HH, 16, k0, lane);
      acc0 = wmma_bf16(a0, bfrag, acc0);
      acc1 = wmma_bf16(a1, bfrag, acc1);
    }
    const int nl   = nt * 16 + (lane & 15);
    const float bi = out_b[nbase + nl];
    const int  mb  = (lane & 16) ? 8 : 0;
#pragma unroll
    for (int r = 0; r < 8; ++r) {
      preds[(mb + r) * 256 + nl]       = acc0[r] + bi;
      preds[(16 + mb + r) * 256 + nl]  = acc1[r] + bi;
    }
  }
  __syncthreads();
  for (int i = tid; i < BB * 256; i += 256) {
    int b = i >> 8, n = i & 255;
    out[(size_t)b * TT * VV + (size_t)t * VV + nbase + n] = preds[i];
  }
  if (tid < BB) {                            // per-block argmax (first-max, ascending n)
    float best = -INFINITY; int bi = nbase;
    for (int n = 0; n < 256; ++n) {
      float v = preds[tid * 256 + n];
      if (v > best) { best = v; bi = nbase + n; }
    }
    part_val[blockIdx.x * BB + tid] = best;
    part_idx[blockIdx.x * BB + tid] = bi;
  }
}

// ---------------- host orchestration ----------------
extern "C" void kernel_launch(void* const* d_in, const int* in_sizes, int n_in,
                              void* d_out, int out_size, void* d_ws, size_t ws_size,
                              hipStream_t stream) {
  const int*   ingredients = (const int*)  d_in[0];
  const float* emb    = (const float*)d_in[1];
  const float* eWih   = (const float*)d_in[2];
  const float* eWhh   = (const float*)d_in[3];
  const float* ebih   = (const float*)d_in[4];
  const float* ebhh   = (const float*)d_in[5];
  const float* dWih   = (const float*)d_in[6];
  const float* dWhh   = (const float*)d_in[7];
  const float* dbih   = (const float*)d_in[8];
  const float* dbhh   = (const float*)d_in[9];
  const float* attn_W = (const float*)d_in[10];
  const float* attn_b = (const float*)d_in[11];
  const float* out_W  = (const float*)d_in[12];
  const float* out_b  = (const float*)d_in[13];
  float* out = (float*)d_out;

  char* w = (char*)d_ws;
  auto alloc = [&](size_t bytes) -> char* {
    char* p = w; w += (bytes + 255) & ~(size_t)255; return p;
  };
  bf16*  emb_bf   = (bf16*)alloc((size_t)VV * EE * 2);
  bf16*  outW_bf  = (bf16*)alloc((size_t)VV * HH * 2);
  bf16*  eWih_bf  = (bf16*)alloc((size_t)GG * EE * 2);
  bf16*  eWhh_bf  = (bf16*)alloc((size_t)GG * HH * 2);
  bf16*  dWih_bf  = (bf16*)alloc((size_t)GG * (EE + HH) * 2);
  bf16*  dWhh_bf  = (bf16*)alloc((size_t)GG * HH * 2);
  bf16*  x_enc    = (bf16*)alloc((size_t)BB * SS * EE * 2);
  float* enc_outs = (float*)alloc((size_t)BB * SS * HH * 4);
  bf16*  ctx_bf   = (bf16*)alloc((size_t)BB * HH * 2);
  float* d_h      = (float*)alloc((size_t)BB * HH * 4);
  bf16*  d_hb     = (bf16*)alloc((size_t)BB * HH * 2);
  float* part_val = (float*)alloc((size_t)(VV / 256) * BB * 4);
  int*   part_idx = (int*)  alloc((size_t)(VV / 256) * BB * 4);

  auto cast = [&](const float* s, bf16* d, int n) {
    k_cast_f32_bf16<<<(n + 255) / 256, 256, 0, stream>>>(s, d, n);
  };
  cast(emb,   emb_bf,  VV * EE);
  cast(out_W, outW_bf, VV * HH);
  cast(eWih,  eWih_bf, GG * EE);
  cast(eWhh,  eWhh_bf, GG * HH);
  cast(dWih,  dWih_bf, GG * (EE + HH));
  cast(dWhh,  dWhh_bf, GG * HH);

  k_gather_src<<<(BB * SS * EE + 255) / 256, 256, 0, stream>>>(ingredients, emb, x_enc);

  // Encoder: persistent single workgroup; LDS = gi+gh (f32) + hb (bf16)
  size_t ldsEnc = (size_t)BB * GG * 4 * 2 + (size_t)BB * HH * 2;   // 212992 B (<320KB/WGP)
  k_enc_gru<<<1, 512, ldsEnc, stream>>>(x_enc, eWih_bf, eWhh_bf, ebih, ebhh,
                                        enc_outs, d_h, d_hb);

  k_attn<<<BB, 128, 0, stream>>>(enc_outs, attn_W, attn_b, ctx_bf);

  // Decoder: 2 kernels per step (argmax + tok-init fused into k_dec_gru head)
  size_t ldsDec = (size_t)BB * GG * 4 * 2 + (size_t)BB * (EE + HH) * 2 + BB * 4; // 221312 B
  for (int t = 0; t < TT; ++t) {
    k_dec_gru<<<1, 512, ldsDec, stream>>>(emb_bf, ctx_bf, part_val, part_idx,
                                          dWih_bf, dWhh_bf, dbih, dbhh, d_h, d_hb, t);
    k_out_gemm<<<VV / 256, 256, 0, stream>>>(d_hb, outW_bf, out_b, out, t,
                                             part_val, part_idx);
  }
}
